// GNNStack_7172595384488
// MI455X (gfx1250) — matrix operand order repaired and checked by reading, not compiled, not run on hardware
//
#include <hip/hip_runtime.h>

// ---------------------------------------------------------------------------
// Bipartite GNN stack (L=2) for MI455X / gfx1250, wave32 + WMMA bf16.
//
// Reference-faithful simplifications (exact in fp32 semantics):
//  * The s2t message pass only populates aggr_t rows [0,N), but x_new keeps
//    xt[N:], which read aggr_t rows [N,2N) == 0. We feed a literal zero block
//    through the full agg_t_W matmul instead (bit-exact), skipping the whole
//    msg_t scatter.
//  * edge_attr after the last layer is unused -> edge update runs for l=0 only.
//
// v2 changes vs v1 (driven by disasm): all WMMA operands now come from LDS as
// pre-converted, pre-padded bf16.  Weights are staged once per block into LDS
// transposed ([N][Kpad], pad rows zeroed) so B-fragments are 8 contiguous
// ds_load_b32 with no divergent guards; gathered activations are converted
// f32->bf16 once at panel-write.  This removes the per-fragment exec-masked
// scalar global loads and the repeated conversion VALU storms seen in v1.
// ---------------------------------------------------------------------------

#define N_NODES 20000
#define TOTAL   40000
#define NEDGE   500000
#define D       64

typedef __attribute__((ext_vector_type(16))) __bf16  v16bf;
typedef __attribute__((ext_vector_type(8)))  float   v8f;

__device__ __forceinline__ unsigned short f2bf(float f) {
    unsigned int u = __float_as_uint(f);
    u += 0x7fffu + ((u >> 16) & 1u);          // round-to-nearest-even
    return (unsigned short)(u >> 16);
}
__device__ __forceinline__ unsigned int pk2(float a, float b) {
    return (unsigned int)f2bf(a) | ((unsigned int)f2bf(b) << 16);
}

union BF16x16 { v16bf v; unsigned int w[8]; };

// A fragment from bf16 LDS row (K contiguous): lane holds row m = lane%16,
// K offsets {b0..b0+7, b0+16..b0+23}, b0 = 8 for lanes 16..31.
// row points at element K=32*c of the lane's panel row; must be 4B aligned.
__device__ __forceinline__ v16bf a_frag_bf(const unsigned short* row, int hi) {
    const unsigned int* p = (const unsigned int*)row;
    const int o = hi ? 4 : 0;
    BF16x16 t;
#pragma unroll
    for (int i = 0; i < 4; i++) t.w[i]     = p[o + i];        // K b0+0..7
#pragma unroll
    for (int i = 0; i < 4; i++) t.w[4 + i] = p[o + 8 + i];    // K b0+16..23
    return t.v;
}

// B fragment from transposed bf16 LDS weights ([N][Kpad]): lane holds col n,
// K = 32*c + {b0..b0+7, b0+16..b0+23}, b0 = hi?8:0.  col points at element
// [n][32*c] of the staged matrix.
__device__ __forceinline__ v16bf b_frag_bf(const unsigned short* colp, int hi) {
    const unsigned int* p = (const unsigned int*)colp;
    const int o = hi ? 4 : 0;
    BF16x16 t;
#pragma unroll
    for (int i = 0; i < 8; i++) t.w[i] = p[o + (i & 3) + ((i >> 2) << 3)];
    return t.v;
}

// Stage global f32 weights W[K][NCOL] (row-major) into LDS bf16 transposed
// dst[n*KPAD + k], zeroing pad rows k >= klim.  Once per block.
template <int KPAD, int NCOL>
__device__ __forceinline__ void stage_w(const float* __restrict__ W, int klim,
                                        unsigned short* dst) {
    for (int flat = threadIdx.x; flat < NCOL * KPAD; flat += (int)blockDim.x) {
        int k = flat % KPAD;
        int n = flat / KPAD;
        unsigned short v = 0;
        if (k < klim) v = f2bf(W[(size_t)k * NCOL + n]);
        dst[flat] = v;
    }
}

#define WMMA_BF16(A, B, C) \
    __builtin_amdgcn_wmma_f32_16x16x32_bf16(false, (A), false, (B), (short)0, (C), false, false)

// ---------------------------------------------------------------------------
// Kernel 1: t2s messages + scatter.  m2 = relu([x[src-N] ‖ ea] @ W + b) for
// edges with (src>=N && tgt<N); atomicAdd into aggr[tgt].  16 edges per wave,
// 8 waves per block.  A panel 16x96 bf16 (80 real K); 4 N-tiles x 3 K-steps.
// ---------------------------------------------------------------------------
__global__ __launch_bounds__(256)
void k_msg_scatter(const float* __restrict__ x, int xld,
                   const float* __restrict__ ea,
                   const int* __restrict__ src, const int* __restrict__ tgt,
                   const float* __restrict__ W,      // [80][64]
                   const float* __restrict__ bias,   // [64]
                   float* __restrict__ aggr)         // [N][64]
{
    __shared__ unsigned short Wsh[64 * 96];           // transposed [n][k], 12 KB
    __shared__ unsigned short Ap[8][16][96];          // bf16 panels, 24 KB
    __shared__ int            tg[8][16];

    const int tid  = threadIdx.x;
    const int wv   = tid >> 5;
    const int wl   = tid & 31;
    const int e16  = wl & 15;
    const int hi   = wl >> 4;
    const int e0   = (blockIdx.x * 8 + wv) * 16;
    const int edge = e0 + e16;

    stage_w<96, 64>(W, 80, Wsh);

    // ---- phase 1: gather [x[src-N] ‖ ea ‖ 0] -> bf16 panel (2 lanes/edge) ----
    {
        bool msk = false; int t_ = -1, s_ = 0;
        if (edge < NEDGE) {
            s_  = src[edge];
            t_  = tgt[edge];
            msk = (s_ >= N_NODES) && (t_ < N_NODES);
        }
        unsigned short* dst = &Ap[wv][e16][0];
        uint2 z2; z2.x = 0u; z2.y = 0u;
        if (msk) {
            const float* xr = x + (size_t)(s_ - N_NODES) * xld + 32 * hi;
#pragma unroll
            for (int j = 0; j < 8; j++) {                       // 32 x-cols per half-lane
                float4 v = ((const float4*)xr)[j];
                uint2 w; w.x = pk2(v.x, v.y); w.y = pk2(v.z, v.w);
                ((uint2*)(dst + 32 * hi))[j] = w;
            }
            if (hi) {
                const float* er = ea + (size_t)edge * 16;
#pragma unroll
                for (int j = 0; j < 4; j++) {                   // ea -> cols 64..79
                    float4 v = ((const float4*)er)[j];
                    uint2 w; w.x = pk2(v.x, v.y); w.y = pk2(v.z, v.w);
                    ((uint2*)(dst + 64))[j] = w;
                }
#pragma unroll
                for (int j = 0; j < 4; j++) ((uint2*)(dst + 80))[j] = z2;  // pad
            }
        } else {
            if (hi == 0) {
#pragma unroll
                for (int j = 0; j < 8; j++)  ((uint2*)dst)[j] = z2;
            } else {
#pragma unroll
                for (int j = 0; j < 16; j++) ((uint2*)(dst + 32))[j] = z2;
            }
        }
        if (hi == 0) tg[wv][e16] = msk ? t_ : -1;
    }
    __syncthreads();

    // ---- phase 2: 16x96 @ 96x64 via WMMA ----
    v16bf a0 = a_frag_bf(&Ap[wv][e16][0],  hi);
    v16bf a1 = a_frag_bf(&Ap[wv][e16][32], hi);
    v16bf a2 = a_frag_bf(&Ap[wv][e16][64], hi);
    const int col = e16;
    v8f acc[4];
#pragma unroll
    for (int t = 0; t < 4; t++) {
        const unsigned short* wc = &Wsh[(16 * t + col) * 96];
        v8f c = {0.f, 0.f, 0.f, 0.f, 0.f, 0.f, 0.f, 0.f};
        c = WMMA_BF16(a0, b_frag_bf(wc,      hi), c);
        c = WMMA_BF16(a1, b_frag_bf(wc + 32, hi), c);
        c = WMMA_BF16(a2, b_frag_bf(wc + 64, hi), c);
        acc[t] = c;
    }

    // ---- phase 3: bias + relu + atomic scatter (segment_sum) ----
#pragma unroll
    for (int t = 0; t < 4; t++) {
        float bn = bias[16 * t + col];
#pragma unroll
        for (int r = 0; r < 8; r++) {
            int mrow = r + (hi ? 8 : 0);
            int tno  = tg[wv][mrow];
            if (tno >= 0) {
                float v = acc[t][r] + bn;
                v = v > 0.f ? v : 0.f;
                atomicAdd(&aggr[(size_t)tno * D + 16 * t + col], v);
            }
        }
    }
}

// ---------------------------------------------------------------------------
// Kernel 2: node update.  Rows < N: l2norm(relu([aggr ‖ x] @ Ws + bs)).
// Rows >= N: l2norm(relu([0 ‖ x] @ Wt + bt))  (the exact dead aggr_t==0 path).
// 16 rows per wave, 4 waves per block; N=20000 is a multiple of 16 so tiles
// never straddle the s/t boundary.  4 N-tiles x 4 K-steps = 16 WMMAs.
// The f32 l2norm staging aliases the bf16 panel region (same 4 KB per wave).
// ---------------------------------------------------------------------------
__global__ __launch_bounds__(128)
void k_node_update(const float* __restrict__ x, int xld,
                   const float* __restrict__ aggr,
                   const float* __restrict__ Ws, const float* __restrict__ bs,
                   const float* __restrict__ Wt, const float* __restrict__ bt,
                   float* __restrict__ out, int outColOff)   // out: [TOTAL][128]
{
    __shared__ unsigned short WshS[64 * 128];                 // 16 KB
    __shared__ unsigned short WshT[64 * 128];                 // 16 KB
    __shared__ __align__(16) unsigned char pan[4][16 * 128 * 2];  // 16 KB

    const int tid  = threadIdx.x;
    const int wv   = tid >> 5;
    const int wl   = tid & 31;
    const int e16  = wl & 15;
    const int hi   = wl >> 4;
    const int row0 = (blockIdx.x * 4 + wv) * 16;
    const int row  = row0 + e16;
    const bool isS = (row0 < N_NODES);

    stage_w<128, 64>(Ws, 128, WshS);
    stage_w<128, 64>(Wt, 128, WshT);

    unsigned short* Apan = (unsigned short*)pan[wv];          // [16][128] bf16

    // ---- phase 1: build [aggr_or_zero ‖ x] bf16 panel ----
    {
        unsigned short* dst = Apan + e16 * 128;
        uint2 z2; z2.x = 0u; z2.y = 0u;
        if (row < TOTAL) {
            if (isS) {
                const float* ar = aggr + (size_t)row * D + 32 * hi;
#pragma unroll
                for (int j = 0; j < 8; j++) {
                    float4 v = ((const float4*)ar)[j];
                    uint2 w; w.x = pk2(v.x, v.y); w.y = pk2(v.z, v.w);
                    ((uint2*)(dst + 32 * hi))[j] = w;
                }
            } else {
#pragma unroll
                for (int j = 0; j < 8; j++) ((uint2*)(dst + 32 * hi))[j] = z2;
            }
            const float* xr = x + (size_t)row * xld + 32 * hi;
#pragma unroll
            for (int j = 0; j < 8; j++) {
                float4 v = ((const float4*)xr)[j];
                uint2 w; w.x = pk2(v.x, v.y); w.y = pk2(v.z, v.w);
                ((uint2*)(dst + 64 + 32 * hi))[j] = w;
            }
        } else {
#pragma unroll
            for (int j = 0; j < 16; j++) ((uint2*)(dst + 64 * hi))[j] = z2;
        }
    }
    __syncthreads();

    // ---- phase 2: 16x128 @ 128x64 via WMMA ----
    const unsigned short* Arow = Apan + e16 * 128;
    v16bf a0 = a_frag_bf(Arow,      hi);
    v16bf a1 = a_frag_bf(Arow + 32, hi);
    v16bf a2 = a_frag_bf(Arow + 64, hi);
    v16bf a3 = a_frag_bf(Arow + 96, hi);
    const unsigned short* Wsh = isS ? WshS : WshT;
    const float* B = isS ? bs : bt;
    const int col = e16;
    v8f acc[4];
#pragma unroll
    for (int t = 0; t < 4; t++) {
        const unsigned short* wc = &Wsh[(16 * t + col) * 128];
        v8f c = {0.f, 0.f, 0.f, 0.f, 0.f, 0.f, 0.f, 0.f};
        c = WMMA_BF16(a0, b_frag_bf(wc,      hi), c);
        c = WMMA_BF16(a1, b_frag_bf(wc + 32, hi), c);
        c = WMMA_BF16(a2, b_frag_bf(wc + 64, hi), c);
        c = WMMA_BF16(a3, b_frag_bf(wc + 96, hi), c);
        acc[t] = c;
    }
    __syncthreads();   // all panel reads done; region is reused as f32 below

    // ---- phase 3: bias + relu, stage f32 into aliased panel region ----
    float* On = (float*)pan[wv];                              // [16][64] f32
#pragma unroll
    for (int t = 0; t < 4; t++) {
        float bn = B[16 * t + col];
#pragma unroll
        for (int r = 0; r < 8; r++) {
            int mrow = r + (hi ? 8 : 0);
            float v = acc[t][r] + bn;
            On[mrow * 64 + 16 * t + col] = v > 0.f ? v : 0.f;
        }
    }
    __syncthreads();

    // ---- phase 4: row l2-norm (2 lanes per row) + store ----
    {
        const float* sp = On + e16 * 64 + 32 * hi;
        float ss = 0.f;
#pragma unroll
        for (int j = 0; j < 32; j++) ss += sp[j] * sp[j];
        ss += __shfl_xor(ss, 16, 32);                  // combine the two halves
        float inv = 1.0f / fmaxf(sqrtf(ss), 1e-12f);
        if (row < TOTAL) {
            float* op = out + (size_t)row * 128 + outColOff + 32 * hi;
#pragma unroll
            for (int j = 0; j < 32; j++) op[j] = sp[j] * inv;
        }
    }
}

// ---------------------------------------------------------------------------
// Kernel 3: edge attr update (layer 0 only; layer 1's ea is unused).
// ea' = relu([x_new[src] ‖ x_new[tgt] ‖ ea] @ upd_W + upd_b).
// A panel 16x160 bf16 (144 real K); 1 N-tile x 5 K-steps = 5 WMMAs / 16 edges.
// ---------------------------------------------------------------------------
__global__ __launch_bounds__(128)
void k_edge_update(const float* __restrict__ xn, int xColOff,  // [TOTAL][128]
                   const float* __restrict__ ea_in,
                   const int* __restrict__ src, const int* __restrict__ tgt,
                   const float* __restrict__ W,      // [144][16]
                   const float* __restrict__ bias,   // [16]
                   float* __restrict__ ea_out)       // [E][16]
{
    __shared__ unsigned short Wsh[16 * 160];          // 5 KB
    __shared__ unsigned short Ap[4][16][160];         // 20 KB

    const int tid  = threadIdx.x;
    const int wv   = tid >> 5;
    const int wl   = tid & 31;
    const int e16  = wl & 15;
    const int hi   = wl >> 4;
    const int e0   = (blockIdx.x * 4 + wv) * 16;
    const int edge = e0 + e16;

    stage_w<160, 16>(W, 144, Wsh);

    {
        unsigned short* dst = &Ap[wv][e16][0];
        uint2 z2; z2.x = 0u; z2.y = 0u;
        if (edge < NEDGE) {
            int s_ = src[edge], t_ = tgt[edge];
            const float* xr = xn + (size_t)(hi ? t_ : s_) * 128 + xColOff;
#pragma unroll
            for (int j = 0; j < 16; j++) {                    // 64 cols per half-lane
                float4 v = ((const float4*)xr)[j];
                uint2 w; w.x = pk2(v.x, v.y); w.y = pk2(v.z, v.w);
                ((uint2*)(dst + 64 * hi))[j] = w;
            }
            if (hi) {
                const float* er = ea_in + (size_t)edge * 16;
#pragma unroll
                for (int j = 0; j < 4; j++) {                 // ea -> cols 128..143
                    float4 v = ((const float4*)er)[j];
                    uint2 w; w.x = pk2(v.x, v.y); w.y = pk2(v.z, v.w);
                    ((uint2*)(dst + 128))[j] = w;
                }
#pragma unroll
                for (int j = 0; j < 4; j++) ((uint2*)(dst + 144))[j] = z2;  // pad
            }
        } else {
#pragma unroll
            for (int j = 0; j < 20; j++) ((uint2*)(dst + 80 * hi))[j] = z2;
        }
    }
    __syncthreads();

    v16bf a[5];
#pragma unroll
    for (int c = 0; c < 5; c++) a[c] = a_frag_bf(&Ap[wv][e16][32 * c], hi);

    const int col = e16;
    const unsigned short* wc = &Wsh[col * 160];
    v8f acc = {0.f, 0.f, 0.f, 0.f, 0.f, 0.f, 0.f, 0.f};
#pragma unroll
    for (int c = 0; c < 5; c++)
        acc = WMMA_BF16(a[c], b_frag_bf(wc + 32 * c, hi), acc);

    float bn = bias[col];
#pragma unroll
    for (int r = 0; r < 8; r++) {
        int mrow = r + (hi ? 8 : 0);
        int eidx = e0 + mrow;
        if (eidx < NEDGE) {
            float v = acc[r] + bn;
            ea_out[(size_t)eidx * 16 + col] = v > 0.f ? v : 0.f;
        }
    }
}

// ---------------------------------------------------------------------------
// Host orchestration.  x_new for each layer is written directly into d_out
// (stride 128); layer 1 reads it back from there.  Workspace: aggr (N*64 f32)
// + updated ea (E*16 f32) = ~37.1 MB.
// ---------------------------------------------------------------------------
extern "C" void kernel_launch(void* const* d_in, const int* in_sizes, int n_in,
                              void* d_out, int out_size, void* d_ws, size_t ws_size,
                              hipStream_t stream) {
    const float* x       = (const float*)d_in[0];
    const float* ea      = (const float*)d_in[1];
    const int*   eidx    = (const int*)  d_in[2];   // [2][E] int32
    const float* msg_s_W = (const float*)d_in[3];
    const float* msg_s_b = (const float*)d_in[4];
    // d_in[5], d_in[6] (msg_t_W/b) are dead code in the reference: their
    // aggregation only reaches xt rows [0,N), which are discarded.
    const float* agg_s_W = (const float*)d_in[7];
    const float* agg_s_b = (const float*)d_in[8];
    const float* agg_t_W = (const float*)d_in[9];
    const float* agg_t_b = (const float*)d_in[10];
    const float* upd_W   = (const float*)d_in[11];
    const float* upd_b   = (const float*)d_in[12];
    float* out = (float*)d_out;

    const int* src = eidx;
    const int* tgt = eidx + NEDGE;

    float* aggr = (float*)d_ws;                                                // N*64 f32
    float* ea1  = (float*)((char*)d_ws + (size_t)N_NODES * D * sizeof(float)); // E*16 f32

    dim3 b1(256), g1((NEDGE + 127) / 128);
    dim3 b2(128), g2((TOTAL + 63) / 64);
    dim3 b3(128), g3((NEDGE + 63) / 64);

    // -------- layer 0 --------
    hipMemsetAsync(aggr, 0, (size_t)N_NODES * D * sizeof(float), stream);
    k_msg_scatter<<<g1, b1, 0, stream>>>(x, D, ea, src, tgt, msg_s_W, msg_s_b, aggr);
    k_node_update<<<g2, b2, 0, stream>>>(x, D, aggr, agg_s_W, agg_s_b, agg_t_W, agg_t_b, out, 0);
    k_edge_update<<<g3, b3, 0, stream>>>(out, 0, ea, src, tgt, upd_W, upd_b, ea1);

    // -------- layer 1 -------- (x_new0 lives in d_out cols 0..63, stride 128)
    hipMemsetAsync(aggr, 0, (size_t)N_NODES * D * sizeof(float), stream);
    k_msg_scatter<<<g1, b1, 0, stream>>>(out, 128, ea1, src, tgt,
                                         msg_s_W + 80 * 64, msg_s_b + 64, aggr);
    k_node_update<<<g2, b2, 0, stream>>>(out, 128, aggr,
                                         agg_s_W + 128 * 64, agg_s_b + 64,
                                         agg_t_W + 128 * 64, agg_t_b + 64, out, 64);
    // edge update after the final layer is unused -> skipped.
}
